// VarianceAdaptor_83949430767776
// MI455X (gfx1250) — matrix operand order repaired and checked by reading, not compiled, not run on hardware
//
#include <hip/hip_runtime.h>
#include <hip/hip_bf16.h>
#include <cstdint>

typedef __attribute__((ext_vector_type(16))) _Float16 v16h;
typedef __attribute__((ext_vector_type(8)))  _Float16 v8h;
typedef __attribute__((ext_vector_type(8)))  float    v8f;

#define AP 272   // LDS pitch (halfs) for activation tile: 544B rows, 16B aligned
#define HP 260   // LDS pitch (floats) for 16-row conv-out chunk
#define MT 4     // m-tiles per WG (64 rows)

// ---------------- CDNA5 async global->LDS copy (16B per lane) ----------------
__device__ __forceinline__ void async_b128(uint32_t lds_byte, unsigned long long gbase, uint32_t goff) {
  // GVS mode: LDS[vdst] = MEM[saddr + vaddr]
  asm volatile("global_load_async_to_lds_b128 %0, %1, %2"
               :: "v"(lds_byte), "v"(goff), "s"(gbase) : "memory");
}
__device__ __forceinline__ void wait_async0() {
#if __has_builtin(__builtin_amdgcn_s_wait_asynccnt)
  __builtin_amdgcn_s_wait_asynccnt(0);
#else
  asm volatile("s_wait_asynccnt 0x0" ::: "memory");
#endif
}

// ---------------- prep: f32 -> f16 convert ----------------
__global__ __launch_bounds__(256) void cvt_kernel(const float* __restrict__ x,
                                                  _Float16* __restrict__ xh) {
  int i = (blockIdx.x * 256 + threadIdx.x) * 4;
  float4 v = *(const float4*)(x + i);
  _Float16 o[4] = {(_Float16)v.x, (_Float16)v.y, (_Float16)v.z, (_Float16)v.w};
  *(uint2*)(xh + i) = *(const uint2*)o;
}

// ---------------- prep: pack conv weights into WMMA B-fragment layout ----------------
// dst[c][kt][nt][lane][e] = w_c[k = kt*32 + 16*(lane>>4) + e][n = nt*16 + (lane&15)]
__global__ __launch_bounds__(256) void packw_kernel(
    const float* __restrict__ w0, const float* __restrict__ w1,
    const float* __restrict__ w2, const float* __restrict__ w3,
    const float* __restrict__ w4, const float* __restrict__ w5,
    _Float16* __restrict__ wpack) {
  int c = blockIdx.y;
  int el = blockIdx.x * 256 + threadIdx.x;            // 0..196607
  const float* w = (c == 0) ? w0 : (c == 1) ? w1 : (c == 2) ? w2
                 : (c == 3) ? w3 : (c == 4) ? w4 : w5;
  int e    = el & 15;
  int lane = (el >> 4) & 31;
  int nt   = (el >> 9) & 15;
  int kt   = el >> 13;                                // 0..23
  int k = kt * 32 + ((lane >> 4) << 4) + e;           // K in [0,768)
  int n = nt * 16 + (lane & 15);                      // N in [0,256)
  wpack[(size_t)c * 196608 + el] = (_Float16)w[k * 256 + n];
}

// ---------------- core: 64-row x 256-col conv(3,256->256) accumulation ----------------
// 8 waves; wave owns 2 n-tiles; 4 m-tiles per wave; B-fragments reused across m-tiles.
__device__ __forceinline__ void conv_accum64(
    const _Float16* __restrict__ inh, const _Float16* __restrict__ wpack,
    const float* __restrict__ bias, _Float16* sA, float* sbias,
    int tid, int l0, v8f (&acc)[MT][2]) {
  int lane = tid & 31, wave = tid >> 5;
  int bstart = l0 & ~1023;                            // batch boundary (L=1024)
  unsigned long long gbase = (unsigned long long)(uintptr_t)inh;
  uint32_t sAbase = (uint32_t)(uintptr_t)sA;

  sbias[tid] = bias[tid];                             // stage bias while DMA runs

  // stage rows l0-1 .. l0+64 (66 rows, zero-pad outside the batch)
  for (int r = wave; r < 66; r += 8) {
    int g = l0 - 1 + r;
    uint32_t laddr = sAbase + (uint32_t)(r * AP * 2) + (uint32_t)lane * 16u;
    if (g >= bstart && g < bstart + 1024) {
      async_b128(laddr, gbase, (uint32_t)g * 512u + (uint32_t)lane * 16u);
    } else {
      uint4 z = {0u, 0u, 0u, 0u};
      *(uint4*)(sA + r * AP + lane * 8) = z;          // disjoint from async rows
    }
  }
  wait_async0();
  __syncthreads();

  int half = lane >> 4, mrow = lane & 15;
  for (int kt = 0; kt < 24; ++kt) {
    int tap = kt >> 3;                                // conv tap 0..2
    int ci0 = (kt & 7) << 5;                          // channel base within tap
    const v16h* wp = (const v16h*)(wpack + (size_t)(((kt * 16 + wave * 2) * 32 + lane) << 4));
    v16h b0 = wp[0];
    v16h b1 = wp[32];                                 // next n-tile: +512 halfs
#pragma unroll
    for (int m = 0; m < MT; ++m) {
      const _Float16* ap = sA + (m * 16 + mrow + tap) * AP + ci0 + (half << 4);
      v8h alo = *(const v8h*)ap;
      v8h ahi = *(const v8h*)(ap + 8);
      v16h a = __builtin_shufflevector(alo, ahi, 0,1,2,3,4,5,6,7,8,9,10,11,12,13,14,15);
      acc[m][0] = __builtin_amdgcn_wmma_f32_16x16x32_f16(false, a, false, b0, (short)0, acc[m][0], false, false);
      acc[m][1] = __builtin_amdgcn_wmma_f32_16x16x32_f16(false, a, false, b1, (short)0, acc[m][1], false, false);
    }
  }
}

// stage one 16-row chunk (bias+ReLU) into hbuf
__device__ __forceinline__ void stage_chunk(const v8f* accm, float* hbuf, const float* sbias,
                                            int lane, int wave) {
  int half = lane >> 4, mrow = lane & 15;
  int n0 = wave * 32 + mrow, n1 = n0 + 16;
  for (int r = 0; r < 8; ++r) {
    int row = r + (half << 3);
    hbuf[row * HP + n0] = fmaxf(accm[0][r] + sbias[n0], 0.f);
    hbuf[row * HP + n1] = fmaxf(accm[1][r] + sbias[n1], 0.f);
  }
}

__device__ __forceinline__ void ln_stats(const float* hbuf, float* psum, float* psum2,
                                         float* smean, float* sinv, int tid) {
  int row = tid >> 4, part = tid & 15;
  float s = 0.f, s2 = 0.f;
  const float* hr = hbuf + row * HP + part * 16;
  for (int j = 0; j < 16; ++j) { float v = hr[j]; s += v; s2 += v * v; }
  psum[tid] = s; psum2[tid] = s2;
  __syncthreads();
  if (part == 0) {
    float a = 0.f, b = 0.f;
    for (int k = 0; k < 16; ++k) { a += psum[row * 16 + k]; b += psum2[row * 16 + k]; }
    float mean = a * (1.f / 256.f);
    float var  = b * (1.f / 256.f) - mean * mean;
    smean[row] = mean; sinv[row] = rsqrtf(var + 1e-5f);
  }
  __syncthreads();
}

// conv + LN, output f16 activations (first layer of each predictor)
__global__ __launch_bounds__(256) void convln_kernel(
    const _Float16* __restrict__ inh, const _Float16* __restrict__ wpack,
    const float* __restrict__ bias, const float* __restrict__ gamma,
    const float* __restrict__ beta, _Float16* __restrict__ outh) {
  __shared__ _Float16 sA[66 * AP];
  __shared__ float hbuf[16 * HP];
  __shared__ float sbias[256], sg[256], sb[256];
  __shared__ float psum[256], psum2[256];
  __shared__ float smean[16], sinv[16];
  int tid = threadIdx.x, lane = tid & 31, wave = tid >> 5;
  int l0 = blockIdx.x * 64;
  sg[tid] = gamma[tid]; sb[tid] = beta[tid];
  v8f acc[MT][2] = {};
  conv_accum64(inh, wpack, bias, sA, sbias, tid, l0, acc);

  int row = tid >> 4, part = tid & 15;
#pragma unroll
  for (int m = 0; m < MT; ++m) {
    stage_chunk(acc[m], hbuf, sbias, lane, wave);
    __syncthreads();
    ln_stats(hbuf, psum, psum2, smean, sinv, tid);
    float mean = smean[row], inv = sinv[row];
    _Float16 o16[16];
    for (int j = 0; j < 16; ++j) {
      int c = part * 16 + j;
      o16[j] = (_Float16)((hbuf[row * HP + c] - mean) * inv * sg[c] + sb[c]);
    }
    _Float16* dst = outh + (size_t)(l0 + m * 16 + row) * 256 + part * 16;
    *(uint4*)dst       = *(const uint4*)&o16[0];
    *(uint4*)(dst + 8) = *(const uint4*)&o16[8];
    __syncthreads();   // hbuf reuse guard for next chunk
  }
}

// conv + LN + linear(256->1) + mask, output f32 predictor values (second layer)
__global__ __launch_bounds__(256) void convln_lin_kernel(
    const _Float16* __restrict__ inh, const _Float16* __restrict__ wpack,
    const float* __restrict__ bias, const float* __restrict__ gamma,
    const float* __restrict__ beta, const float* __restrict__ lw,
    const float* __restrict__ lb, const unsigned char* __restrict__ mask,
    float* __restrict__ outp) {
  __shared__ _Float16 sA[66 * AP];
  __shared__ float hbuf[16 * HP];
  __shared__ float sbias[256], sg[256], sb[256], slw[256];
  __shared__ float psum[256], psum2[256], pdot[256];
  __shared__ float smean[16], sinv[16];
  int tid = threadIdx.x, lane = tid & 31, wave = tid >> 5;
  int l0 = blockIdx.x * 64;
  sg[tid] = gamma[tid]; sb[tid] = beta[tid]; slw[tid] = lw[tid];
  v8f acc[MT][2] = {};
  conv_accum64(inh, wpack, bias, sA, sbias, tid, l0, acc);

  int row = tid >> 4, part = tid & 15;
#pragma unroll
  for (int m = 0; m < MT; ++m) {
    stage_chunk(acc[m], hbuf, sbias, lane, wave);
    __syncthreads();
    ln_stats(hbuf, psum, psum2, smean, sinv, tid);
    float mean = smean[row], inv = sinv[row], dp = 0.f;
    for (int j = 0; j < 16; ++j) {
      int c = part * 16 + j;
      float v = (hbuf[row * HP + c] - mean) * inv * sg[c] + sb[c];
      dp += v * slw[c];
    }
    pdot[tid] = dp;
    __syncthreads();
    if (part == 0) {
      float s = 0.f;
      for (int k = 0; k < 16; ++k) s += pdot[row * 16 + k];
      s += lb[0];
      int g = l0 + m * 16 + row;
      outp[g] = mask[g] ? 0.f : s;
    }
    __syncthreads();   // hbuf/pdot reuse guard for next chunk
  }
}

// ---------------- cumsum of durations (per batch row) ----------------
__global__ __launch_bounds__(256) void cumsum_kernel(
    const int* __restrict__ dur, const int* __restrict__ maxlen,
    int* __restrict__ cum, float* __restrict__ mel_len_out) {
  __shared__ int ssc[256];
  int b = blockIdx.x, t = threadIdx.x;
  int base = b * 1024 + t * 4;
  int d0 = dur[base], d1 = dur[base + 1], d2 = dur[base + 2], d3 = dur[base + 3];
  int s = d0 + d1 + d2 + d3;
  ssc[t] = s; __syncthreads();
  for (int off = 1; off < 256; off <<= 1) {
    int v = (t >= off) ? ssc[t - off] : 0;
    __syncthreads();
    ssc[t] += v;
    __syncthreads();
  }
  int excl = ssc[t] - s;
  int c0 = excl + d0, c1 = c0 + d1, c2 = c1 + d2, c3 = c2 + d3;
  cum[base] = c0; cum[base + 1] = c1; cum[base + 2] = c2; cum[base + 3] = c3;
  if (t == 255) mel_len_out[b] = (float)min(ssc[255], maxlen[0]);
}

// ---------------- bucketize + embedding add + length regulate ----------------
__global__ __launch_bounds__(256) void expand_kernel(
    const float* __restrict__ x, const float* __restrict__ pt,
    const float* __restrict__ et, const float* __restrict__ pbins,
    const float* __restrict__ ebins, const float* __restrict__ pemb,
    const float* __restrict__ eemb, const int* __restrict__ cum,
    const int* __restrict__ maxlen, float* __restrict__ out_exp,
    float* __restrict__ out_mask) {
  __shared__ int sidx[16], spi[16], sei[16], sval[16];
  int bx = blockIdx.x;
  int b = bx >> 8;
  int t0 = (bx & 255) * 16;
  int tid = threadIdx.x;
  if (tid < 16) {
    int t = t0 + tid;
    const int* c = cum + b * 1024;
    int mel_len = min(c[1023], maxlen[0]);
    int lo = 0, hi = 1024;                    // searchsorted(cum, t, 'right')
    while (lo < hi) { int mid = (lo + hi) >> 1; if (c[mid] <= t) lo = mid + 1; else hi = mid; }
    int idx = min(lo, 1023);
    sidx[tid] = idx;
    int valid = (t < mel_len);
    sval[tid] = valid;
    float pv = pt[b * 1024 + idx], ev = et[b * 1024 + idx];
    lo = 0; hi = 255;                         // searchsorted(bins, v, 'left')
    while (lo < hi) { int mid = (lo + hi) >> 1; if (pbins[mid] < pv) lo = mid + 1; else hi = mid; }
    spi[tid] = lo;
    lo = 0; hi = 255;
    while (lo < hi) { int mid = (lo + hi) >> 1; if (ebins[mid] < ev) lo = mid + 1; else hi = mid; }
    sei[tid] = lo;
    out_mask[b * 4096 + t] = valid ? 0.f : 1.f;
  }
  __syncthreads();
  for (int r = 0; r < 16; ++r) {
    float v = 0.f;
    if (sval[r]) {
      v = x[(size_t)(b * 1024 + sidx[r]) * 256 + tid]
        + pemb[(size_t)spi[r] * 256 + tid]
        + eemb[(size_t)sei[r] * 256 + tid];
    }
    out_exp[((size_t)(b * 4096 + t0 + r)) * 256 + tid] = v;
  }
}

extern "C" void kernel_launch(void* const* d_in, const int* in_sizes, int n_in,
                              void* d_out, int out_size, void* d_ws, size_t ws_size,
                              hipStream_t stream) {
  const float* x              = (const float*)d_in[0];
  const unsigned char* mask   = (const unsigned char*)d_in[1];
  const int* dur              = (const int*)d_in[2];
  const float* pt             = (const float*)d_in[3];
  const float* et             = (const float*)d_in[4];
  const int* maxlen           = (const int*)d_in[5];
  auto P = [&](int i) { return (const float*)d_in[i]; };

  char* ws = (char*)d_ws;
  _Float16* xh    = (_Float16*)ws;                       // 16 MB: x in f16
  _Float16* h1    = (_Float16*)(ws + 16777216);          // 16 MB: intermediate acts
  _Float16* wpack = (_Float16*)(ws + 33554432);          // 2.25 MB: packed weights
  int* cum        = (int*)(ws + 35913728);               // 128 KB

  float* out      = (float*)d_out;
  float* out_exp  = out;                                 // [32,4096,256]
  float* out_dur  = out + 33554432;                      // [32,1024]
  float* out_pit  = out_dur + 32768;
  float* out_en   = out_pit + 32768;
  float* out_mlen = out_en + 32768;                      // [32]
  float* out_mask = out_mlen + 32;                       // [32,4096]

  cvt_kernel<<<8192, 256, 0, stream>>>(x, xh);
  packw_kernel<<<dim3(768, 6), 256, 0, stream>>>(P(6), P(10), P(16), P(20), P(26), P(30), wpack);

  const int pbase[3] = {6, 16, 26};
  float* pout[3] = {out_dur, out_pit, out_en};
  for (int p = 0; p < 3; ++p) {
    int B0 = pbase[p];
    convln_kernel<<<512, 256, 0, stream>>>(
        xh, wpack + (size_t)(p * 2) * 196608, P(B0 + 1), P(B0 + 2), P(B0 + 3), h1);
    convln_lin_kernel<<<512, 256, 0, stream>>>(
        h1, wpack + (size_t)(p * 2 + 1) * 196608, P(B0 + 5), P(B0 + 6), P(B0 + 7),
        P(B0 + 8), P(B0 + 9), mask, pout[p]);
  }

  cumsum_kernel<<<32, 256, 0, stream>>>(dur, maxlen, cum, out_mlen);
  expand_kernel<<<8192, 256, 0, stream>>>(x, pt, et, P(36), P(37), P(38), P(39),
                                          cum, maxlen, out_exp, out_mask);
}
